// AttentionBlock_67937792688673
// MI455X (gfx1250) — matrix operand order repaired and checked by reading, not compiled, not run on hardware
//
#include <hip/hip_runtime.h>

#define C_DIM 512
#define NSEQ  1024
#define BATCH 8
#define HEADS 8
#define HD    64
#define GROUPS 32

typedef __bf16 bf16_t;
typedef __attribute__((ext_vector_type(16))) __bf16 v16bf;
typedef __attribute__((ext_vector_type(8)))  __bf16 v8bf;
typedef __attribute__((ext_vector_type(8)))  float  v8f;

static __device__ inline v16bf make16(v8bf lo, v8bf hi) {
  v16bf r;
#pragma unroll
  for (int i = 0; i < 8; ++i) { r[i] = lo[i]; r[i + 8] = hi[i]; }
  return r;
}

// A operand (16x32 bf16, MxK) from row-major [M][K] memory (ld = row stride in elems).
// ISA layout: lanes 0-15 row M=lane, K {0..7,16..23}; lanes 16-31 row M=lane-16, K {8..15,24..31}.
static __device__ inline v16bf load_a32(const bf16_t* base, int ld, int lane) {
  int r = lane & 15, hi = lane >> 4;
  const bf16_t* p = base + (size_t)r * ld + 8 * hi;
  v8bf lo = *(const v8bf*)p;
  v8bf h2 = *(const v8bf*)(p + 16);
  return make16(lo, h2);
}

// B operand (32x16 bf16, KxN) supplied as B^T row-major [N][K].
// ISA layout: lane&15 = column n; lanes 0-15 hold K {0..7,16..23}, lanes 16-31 hold K {8..15,24..31}.
static __device__ inline v16bf load_b32(const bf16_t* base, int ld, int lane) {
  int r = lane & 15, hi = lane >> 4;
  const bf16_t* p = base + (size_t)r * ld + 8 * hi;
  v8bf lo = *(const v8bf*)p;
  v8bf h2 = *(const v8bf*)(p + 16);
  return make16(lo, h2);
}

static __device__ inline v8f wmma_bf16(v16bf a, v16bf b, v8f c) {
  // 8 args: (neg_a, A, neg_b, B, c_mod, C, reuse_a, reuse_b)
  return __builtin_amdgcn_wmma_f32_16x16x32_bf16(false, a, false, b, (short)0, c,
                                                 false, false);
}

static __device__ inline v8f zero8() {
  v8f z = {0.f, 0.f, 0.f, 0.f, 0.f, 0.f, 0.f, 0.f};
  return z;
}

// ---------------------------------------------------------------- f32 -> bf16
__global__ __launch_bounds__(256) void cvt_bf16_kernel(const float* __restrict__ in,
                                                       bf16_t* __restrict__ out, int n) {
  int i = blockIdx.x * 256 + threadIdx.x;
  if (i < n) out[i] = (__bf16)in[i];
}

// ---------------------------------------------------------------- GroupNorm
// One block per (b, group). Group = 16 channels x 1024 spatial = 16384 elems.
// Writes h^T bf16 as [b][n][c] (K-major for qkv GEMM B operand).
__global__ __launch_bounds__(256) void gn_kernel(const float* __restrict__ x,
                                                 const float* __restrict__ scale,
                                                 const float* __restrict__ bias,
                                                 bf16_t* __restrict__ hT) {
  int b = blockIdx.x >> 5;
  int g = blockIdx.x & 31;
  const float* xg = x + ((size_t)b * C_DIM + g * 16) * NSEQ;

  float s = 0.f, s2 = 0.f;
  for (int i = threadIdx.x; i < 16 * NSEQ; i += 256) {
    float v = xg[i];
    s += v; s2 += v * v;
  }
#pragma unroll
  for (int off = 16; off >= 1; off >>= 1) {
    s  += __shfl_xor(s,  off, 32);
    s2 += __shfl_xor(s2, off, 32);
  }
  __shared__ float red0[8], red1[8];
  int wave = threadIdx.x >> 5, lane = threadIdx.x & 31;
  if (lane == 0) { red0[wave] = s; red1[wave] = s2; }
  __syncthreads();
  s = 0.f; s2 = 0.f;
#pragma unroll
  for (int w = 0; w < 8; ++w) { s += red0[w]; s2 += red1[w]; }
  float mean = s * (1.f / 16384.f);
  float var  = s2 * (1.f / 16384.f) - mean * mean;
  float rstd = rsqrtf(var + 1e-5f);

  for (int i = threadIdx.x; i < 16 * NSEQ; i += 256) {
    int cl = i >> 10;          // channel within group
    int n  = i & 1023;         // spatial
    int c  = g * 16 + cl;
    float hv = (xg[i] - mean) * rstd * scale[c] + bias[c];
    hT[((size_t)b * NSEQ + n) * C_DIM + c] = (__bf16)hv;
  }
}

// ---------------------------------------------------------------- qkv GEMM
// D[o,n] = sum_c W[o,c] * h[c,n] + b[o]. A = W (row-major, K-major), B = hT [n][c].
// Writes qT (pre-scaled by 1/sqrt(hd)), kT as [bh][n][64]; v as [bh][64][n].
__global__ __launch_bounds__(128) void qkv_kernel(const bf16_t* __restrict__ W,
                                                  const float* __restrict__ bias,
                                                  const bf16_t* __restrict__ hT,
                                                  bf16_t* __restrict__ qT,
                                                  bf16_t* __restrict__ kT,
                                                  bf16_t* __restrict__ vM) {
  int b    = blockIdx.z;
  int lane = threadIdx.x & 31;
  int o0   = blockIdx.x * 64 + (threadIdx.x >> 5) * 16;
  int nB   = blockIdx.y * 64;
  const bf16_t* hB = hT + (size_t)b * NSEQ * C_DIM;

  v8f acc[4];
#pragma unroll
  for (int s = 0; s < 4; ++s) acc[s] = zero8();

  for (int k0 = 0; k0 < C_DIM; k0 += 32) {
    v16bf a = load_a32(W + (size_t)o0 * C_DIM + k0, C_DIM, lane);
#pragma unroll
    for (int s = 0; s < 4; ++s) {
      v16bf bb = load_b32(hB + (size_t)(nB + s * 16) * C_DIM + k0, C_DIM, lane);
      acc[s] = wmma_bf16(a, bb, acc[s]);
    }
  }

  int col = lane & 15, hi = lane >> 4;
#pragma unroll
  for (int s = 0; s < 4; ++s) {
    int n = nB + s * 16 + col;
#pragma unroll
    for (int r = 0; r < 8; ++r) {
      int o = o0 + r + 8 * hi;
      float val = acc[s][r] + bias[o];
      int sec  = o >> 9;        // 0=q 1=k 2=v
      int oc   = o & 511;
      int head = oc >> 6, cl = oc & 63;
      size_t bh = (size_t)(b * HEADS + head);
      if (sec == 0)
        qT[(bh * NSEQ + n) * HD + cl] = (__bf16)(val * 0.125f);   // 1/sqrt(64)
      else if (sec == 1)
        kT[(bh * NSEQ + n) * HD + cl] = (__bf16)val;
      else
        vM[(bh * HD + cl) * NSEQ + n] = (__bf16)val;
    }
  }
}

// ---------------------------------------------------------------- fused attention
// Grid (B*HEADS, NSEQ/64), 4 waves; each wave owns 16 query rows, flash loop
// over 64-wide key/value blocks with online softmax.
__global__ __launch_bounds__(128) void attn_kernel(const bf16_t* __restrict__ qT,
                                                   const bf16_t* __restrict__ kT,
                                                   const bf16_t* __restrict__ vM,
                                                   bf16_t* __restrict__ outT) {
  int bh   = blockIdx.x;
  int b    = bh / HEADS, head = bh % HEADS;
  int wave = threadIdx.x >> 5;
  int lane = threadIdx.x & 31;
  int n0   = blockIdx.y * 64 + wave * 16;

  const bf16_t* q  = qT + (size_t)bh * NSEQ * HD;  // [n][64]
  const bf16_t* kk = kT + (size_t)bh * NSEQ * HD;  // [m][64]
  const bf16_t* vh = vM + (size_t)bh * HD * NSEQ;  // [c][m]

  __shared__ __align__(16) __bf16 ldsP[4][16][64];  // per-wave P tile

  // Q strip: rows n0..n0+15, K = 64 as two K=32 chunks (A operand, held in regs)
  v16bf aq[2];
  aq[0] = load_a32(q + (size_t)n0 * HD +  0, HD, lane);
  aq[1] = load_a32(q + (size_t)n0 * HD + 32, HD, lane);

  v8f O[4];
#pragma unroll
  for (int s = 0; s < 4; ++s) O[s] = zero8();
  float mrow[8], lrow[8];
#pragma unroll
  for (int r = 0; r < 8; ++r) { mrow[r] = -1e30f; lrow[r] = 0.f; }

  for (int m0 = 0; m0 < NSEQ; m0 += 64) {
    // ---- S = (q/sqrt(d)) . k^T  : 16 x 64 strip, 8 WMMAs
    v8f S[4];
#pragma unroll
    for (int s = 0; s < 4; ++s) {
      v16bf bk0 = load_b32(kk + (size_t)(m0 + s * 16) * HD +  0, HD, lane);
      v16bf bk1 = load_b32(kk + (size_t)(m0 + s * 16) * HD + 32, HD, lane);
      v8f acc = zero8();
      acc = wmma_bf16(aq[0], bk0, acc);
      acc = wmma_bf16(aq[1], bk1, acc);
      S[s] = acc;
    }

    // ---- online softmax: rows are (r + 8*hi); reduce over 16 lanes (columns)
    float alpha[8];
#pragma unroll
    for (int r = 0; r < 8; ++r) {
      float mx = fmaxf(fmaxf(S[0][r], S[1][r]), fmaxf(S[2][r], S[3][r]));
#pragma unroll
      for (int off = 1; off < 16; off <<= 1)
        mx = fmaxf(mx, __shfl_xor(mx, off, 16));
      float mnew = fmaxf(mrow[r], mx);
      alpha[r] = __expf(mrow[r] - mnew);
      mrow[r] = mnew;
      float rs = 0.f;
#pragma unroll
      for (int s = 0; s < 4; ++s) {
        float p = __expf(S[s][r] - mnew);
        S[s][r] = p;
        rs += p;
      }
#pragma unroll
      for (int off = 1; off < 16; off <<= 1)
        rs += __shfl_xor(rs, off, 16);
      lrow[r] = lrow[r] * alpha[r] + rs;
    }

    // ---- rescale O; stage P (bf16) into per-wave LDS in [row][m] layout
    int col = lane & 15, hi = lane >> 4;
#pragma unroll
    for (int s = 0; s < 4; ++s) {
#pragma unroll
      for (int r = 0; r < 8; ++r) {
        O[s][r] *= alpha[r];
        ldsP[wave][r + 8 * hi][s * 16 + col] = (__bf16)S[s][r];
      }
    }
    asm volatile("s_wait_dscnt 0" ::: "memory");  // wave-private LDS RAW

    // ---- re-read P in A-operand layout (two K=32 chunks over m)
    v16bf ap[2];
#pragma unroll
    for (int ch = 0; ch < 2; ++ch) {
      const __bf16* p0 = &ldsP[wave][lane & 15][ch * 32 + 8 * hi];
      v8bf lo = *(const v8bf*)p0;
      v8bf h2 = *(const v8bf*)(p0 + 16);
      ap[ch] = make16(lo, h2);
    }

    // ---- O += P . V^T : 8 WMMAs (B operand from v [c][m], K-major over m)
#pragma unroll
    for (int cs = 0; cs < 4; ++cs) {
      v16bf bv0 = load_b32(vh + (size_t)(cs * 16) * NSEQ + m0,      NSEQ, lane);
      v16bf bv1 = load_b32(vh + (size_t)(cs * 16) * NSEQ + m0 + 32, NSEQ, lane);
      O[cs] = wmma_bf16(ap[0], bv0, O[cs]);
      O[cs] = wmma_bf16(ap[1], bv1, O[cs]);
    }
  }

  // ---- normalize and write out^T [b][n][head*64 + c] bf16
#pragma unroll
  for (int cs = 0; cs < 4; ++cs) {
    int c = head * HD + cs * 16 + (lane & 15);
#pragma unroll
    for (int r = 0; r < 8; ++r) {
      int n = n0 + r + 8 * (lane >> 4);
      outT[((size_t)b * NSEQ + n) * C_DIM + c] = (__bf16)(O[cs][r] / lrow[r]);
    }
  }
}

// ---------------------------------------------------------------- proj + residual
__global__ __launch_bounds__(128) void proj_kernel(const bf16_t* __restrict__ P,
                                                   const float* __restrict__ bias,
                                                   const bf16_t* __restrict__ outT,
                                                   const float* __restrict__ x,
                                                   float* __restrict__ out) {
  int b    = blockIdx.z;
  int lane = threadIdx.x & 31;
  int o0   = blockIdx.x * 64 + (threadIdx.x >> 5) * 16;
  int nB   = blockIdx.y * 64;
  const bf16_t* oB = outT + (size_t)b * NSEQ * C_DIM;

  v8f acc[4];
#pragma unroll
  for (int s = 0; s < 4; ++s) acc[s] = zero8();

  for (int k0 = 0; k0 < C_DIM; k0 += 32) {
    v16bf a = load_a32(P + (size_t)o0 * C_DIM + k0, C_DIM, lane);
#pragma unroll
    for (int s = 0; s < 4; ++s) {
      v16bf bb = load_b32(oB + (size_t)(nB + s * 16) * C_DIM + k0, C_DIM, lane);
      acc[s] = wmma_bf16(a, bb, acc[s]);
    }
  }

  int col = lane & 15, hi = lane >> 4;
#pragma unroll
  for (int s = 0; s < 4; ++s) {
    int n = nB + s * 16 + col;
#pragma unroll
    for (int r = 0; r < 8; ++r) {
      int o = o0 + r + 8 * hi;
      size_t idx = ((size_t)b * C_DIM + o) * NSEQ + n;
      out[idx] = x[idx] + bias[o] + acc[s][r];
    }
  }
}

// ---------------------------------------------------------------- launch
extern "C" void kernel_launch(void* const* d_in, const int* in_sizes, int n_in,
                              void* d_out, int out_size, void* d_ws, size_t ws_size,
                              hipStream_t stream) {
  const float* x      = (const float*)d_in[0];
  const float* gscale = (const float*)d_in[1];
  const float* gbias  = (const float*)d_in[2];
  const float* qkvw   = (const float*)d_in[3];
  const float* qkvb   = (const float*)d_in[4];
  const float* projw  = (const float*)d_in[5];
  const float* projb  = (const float*)d_in[6];
  float* out = (float*)d_out;

  // workspace layout (bf16 buffers, 256B-aligned offsets); total ~42 MB
  char* ws = (char*)d_ws;
  bf16_t* wqkv  = (bf16_t*)(ws + 0);                      // 1536*512
  bf16_t* wproj = (bf16_t*)(ws + 1572864);                //  512*512
  bf16_t* hT    = (bf16_t*)(ws + 2097152);                // [B][N][C]
  bf16_t* qT    = (bf16_t*)(ws + 10485760);               // [B*H][N][64]
  bf16_t* kT    = (bf16_t*)(ws + 18874368);               // [B*H][N][64]
  bf16_t* vM    = (bf16_t*)(ws + 27262976);               // [B*H][64][N]
  bf16_t* outT  = (bf16_t*)(ws + 35651584);               // [B][N][C]

  cvt_bf16_kernel<<<(1536 * 512 + 255) / 256, 256, 0, stream>>>(qkvw, wqkv, 1536 * 512);
  cvt_bf16_kernel<<<(512 * 512 + 255) / 256, 256, 0, stream>>>(projw, wproj, 512 * 512);

  gn_kernel<<<BATCH * GROUPS, 256, 0, stream>>>(x, gscale, gbias, hT);

  qkv_kernel<<<dim3(1536 / 64, NSEQ / 64, BATCH), 128, 0, stream>>>(
      wqkv, qkvb, hT, qT, kT, vM);

  attn_kernel<<<dim3(BATCH * HEADS, NSEQ / 64), 128, 0, stream>>>(qT, kT, vM, outT);

  proj_kernel<<<dim3(C_DIM / 64, NSEQ / 64, BATCH), 128, 0, stream>>>(
      wproj, projb, outT, x, out);
}
